// VQCodebook_15685220565597
// MI455X (gfx1250) — matrix operand order repaired and checked by reading, not compiled, not run on hardware
//
#include <hip/hip_runtime.h>

typedef _Float16 v16h __attribute__((ext_vector_type(16)));
typedef _Float16 v8h  __attribute__((ext_vector_type(8)));
typedef float    v8f  __attribute__((ext_vector_type(8)));
typedef float    v4f  __attribute__((ext_vector_type(4)));

#define CODES 1024
#define DIM 64
#define ROWP 72            // padded halves per codebook row in LDS (144B, 16B aligned, bank-spread)
#define WAVES 8
#define TILES_PER_WAVE 2   // 2 x 16 tokens per wave
#define TOKENS_PER_WG (WAVES * TILES_PER_WAVE * 16)  // 256
#define IDXMASK 0xFFFFFC00u   // clear low 10 mantissa bits -> room for 1024 code ids

// d = ||c||^2 - 2 z.c  (||z||^2 constant per token -> irrelevant for argmin)
__global__ __launch_bounds__(256) void vq_wmma_kernel(
    const float* __restrict__ z, const float* __restrict__ cb,
    float* __restrict__ out_zq, float* __restrict__ out_idx,
    float* __restrict__ partial, int n_tokens)
{
    extern __shared__ char smem[];
    _Float16* cbl  = (_Float16*)smem;                         // CODES * ROWP halves (f16 codebook)
    float*    cnrm = (float*)(smem + CODES * ROWP * 2);       // CODES floats (||c||^2)
    float*    wsum = cnrm + CODES;                            // WAVES floats (loss partials)

    const int tid = threadIdx.x;

    // ---- Stage codebook into LDS as f16 (row-major padded) + f32 norms ----
    for (int r = 0; r < CODES / 256; ++r) {
        int code = r * 256 + tid;
        const v4f* src = (const v4f*)(cb + code * DIM);
        _Float16* dst = cbl + code * ROWP;
        float nrm = 0.f;
        #pragma unroll
        for (int q = 0; q < DIM / 4; ++q) {
            v4f f = src[q];
            nrm += f.x * f.x + f.y * f.y + f.z * f.z + f.w * f.w;
            dst[q * 4 + 0] = (_Float16)f.x;
            dst[q * 4 + 1] = (_Float16)f.y;
            dst[q * 4 + 2] = (_Float16)f.z;
            dst[q * 4 + 3] = (_Float16)f.w;
        }
        cnrm[code] = nrm;
    }
    __syncthreads();

    const int lane = tid & 31;
    const int wave = tid >> 5;
    const int c = lane & 15;   // column index within 16-wide tile
    const int h = lane >> 4;   // half-wave id

    float lsum = 0.f;

    for (int it = 0; it < TILES_PER_WAVE; ++it) {
        int token0 = blockIdx.x * TOKENS_PER_WG + wave * (16 * TILES_PER_WAVE) + it * 16;
        if (token0 >= n_tokens) break;

        // ---- Build A-tiles (16 tokens x 32 dims each, contiguous k = 16h + s order) ----
        // lane row m = c; K group: [16h..16h+15] -> a[0..15]
        v16h a[2];
        #pragma unroll
        for (int kk = 0; kk < 2; ++kk) {
            const float* zr = z + (size_t)(token0 + c) * DIM + kk * 32 + 16 * h;
            v4f f0 = *(const v4f*)(zr + 0);
            v4f f1 = *(const v4f*)(zr + 4);
            v4f f2 = *(const v4f*)(zr + 8);
            v4f f3 = *(const v4f*)(zr + 12);
            v16h av;
            #pragma unroll
            for (int i = 0; i < 4; ++i) {
                av[i]      = (_Float16)f0[i];
                av[4 + i]  = (_Float16)f1[i];
                av[8 + i]  = (_Float16)f2[i];
                av[12 + i] = (_Float16)f3[i];
            }
            a[kk] = av;
        }

        // running packed keys: f32 distance with code id in low 10 bits
        float mkey[8];
        #pragma unroll
        for (int j = 0; j < 8; ++j) mkey[j] = 3.4e38f;

        // ---- Sweep 64 code tiles, 2 per iteration (hides WMMA->VALU hazard) ----
        for (int tile = 0; tile < CODES / 16; tile += 2) {
            int code0 = tile * 16 + c;
            int code1 = code0 + 16;
            const _Float16* brow0 = cbl + code0 * ROWP + 16 * h;
            const _Float16* brow1 = cbl + code1 * ROWP + 16 * h;

            v16h b00, b01, b10, b11;
            {
                v8h p0 = *(const v8h*)(brow0 + 0);
                v8h p1 = *(const v8h*)(brow0 + 8);
                v8h p2 = *(const v8h*)(brow0 + 32);
                v8h p3 = *(const v8h*)(brow0 + 40);
                v8h q0 = *(const v8h*)(brow1 + 0);
                v8h q1 = *(const v8h*)(brow1 + 8);
                v8h q2 = *(const v8h*)(brow1 + 32);
                v8h q3 = *(const v8h*)(brow1 + 40);
                #pragma unroll
                for (int i = 0; i < 8; ++i) {
                    b00[i] = p0[i]; b00[8 + i] = p1[i];
                    b01[i] = p2[i]; b01[8 + i] = p3[i];
                    b10[i] = q0[i]; b10[8 + i] = q1[i];
                    b11[i] = q2[i]; b11[8 + i] = q3[i];
                }
            }

            v8f acc0 = {};
            v8f acc1 = {};
            acc0 = __builtin_amdgcn_wmma_f32_16x16x32_f16(false, a[0], false, b00, (short)0, acc0, false, false);
            acc0 = __builtin_amdgcn_wmma_f32_16x16x32_f16(false, a[1], false, b01, (short)0, acc0, false, false);
            acc1 = __builtin_amdgcn_wmma_f32_16x16x32_f16(false, a[0], false, b10, (short)0, acc1, false, false);
            acc1 = __builtin_amdgcn_wmma_f32_16x16x32_f16(false, a[1], false, b11, (short)0, acc1, false, false);

            float cn0 = cnrm[code0];
            float cn1 = cnrm[code1];
            #pragma unroll
            for (int j = 0; j < 8; ++j) {
                float d0 = __builtin_fmaf(-2.0f, acc0[j], cn0);
                float k0 = __uint_as_float((__float_as_uint(d0) & IDXMASK) | (unsigned)code0);
                mkey[j] = fminf(mkey[j], k0);
            }
            #pragma unroll
            for (int j = 0; j < 8; ++j) {
                float d1 = __builtin_fmaf(-2.0f, acc1[j], cn1);
                float k1 = __uint_as_float((__float_as_uint(d1) & IDXMASK) | (unsigned)code1);
                mkey[j] = fminf(mkey[j], k1);
            }
        }

        // ---- Butterfly min across the 16 code-columns (stays within half-wave) ----
        #pragma unroll
        for (int m = 1; m < 16; m <<= 1) {
            #pragma unroll
            for (int j = 0; j < 8; ++j)
                mkey[j] = fminf(mkey[j], __shfl_xor(mkey[j], m, 32));
        }

        // ---- Gather z_q rows (f32 from global codebook), write indices, loss partial ----
        #pragma unroll
        for (int j = 0; j < 8; ++j) {
            int token = token0 + h * 8 + j;
            int idx = (int)(__float_as_uint(mkey[j]) & 1023u);
            v4f cv = *(const v4f*)(cb + (size_t)idx * DIM + 4 * c);
            *(v4f*)(out_zq + (size_t)token * DIM + 4 * c) = cv;
            v4f zv = *(const v4f*)(z + (size_t)token * DIM + 4 * c);
            float dx = cv.x - zv.x, dy = cv.y - zv.y, dz = cv.z - zv.z, dw = cv.w - zv.w;
            lsum += dx * dx + dy * dy + dz * dz + dw * dw;
            if (c == 0) out_idx[token] = (float)idx;
        }
    }

    // ---- Deterministic loss reduction: wave -> LDS -> block partial ----
    #pragma unroll
    for (int m = 1; m < 32; m <<= 1) lsum += __shfl_xor(lsum, m, 32);
    if (lane == 0) wsum[wave] = lsum;
    __syncthreads();
    if (tid == 0) {
        float s = 0.f;
        #pragma unroll
        for (int w = 0; w < WAVES; ++w) s += wsum[w];
        partial[blockIdx.x] = s;
    }
}

__global__ void vq_finalize_kernel(const float* __restrict__ partial, int nblocks,
                                   float* __restrict__ loss_out, float scale)
{
    if (threadIdx.x == 0 && blockIdx.x == 0) {
        float s = 0.f;
        for (int i = 0; i < nblocks; ++i) s += partial[i];   // fixed order: deterministic
        *loss_out = s * scale;
    }
}

extern "C" void kernel_launch(void* const* d_in, const int* in_sizes, int n_in,
                              void* d_out, int out_size, void* d_ws, size_t ws_size,
                              hipStream_t stream) {
    const float* z  = (const float*)d_in[0];   // [8,8192,64] f32
    const float* cb = (const float*)d_in[1];   // [1024,64]   f32

    int n_tokens = in_sizes[0] / DIM;          // 65536

    float* out     = (float*)d_out;
    float* out_zq  = out;                                     // [n_tokens, 64]
    float* out_idx = out + (size_t)n_tokens * DIM;            // [n_tokens]
    float* out_ls  = out + (size_t)n_tokens * DIM + n_tokens; // scalar

    float* partial = (float*)d_ws;

    int nblocks = (n_tokens + TOKENS_PER_WG - 1) / TOKENS_PER_WG;  // 256
    size_t smem = (size_t)CODES * ROWP * 2 + (size_t)CODES * 4 + WAVES * 4;

    vq_wmma_kernel<<<nblocks, 256, smem, stream>>>(z, cb, out_zq, out_idx, partial, n_tokens);

    float scale = 2.0f / (float)((size_t)n_tokens * DIM);  // (1+BETA)/N
    vq_finalize_kernel<<<1, 32, 0, stream>>>(partial, nblocks, out_ls, scale);
}
